// Fusion_4157528343190
// MI455X (gfx1250) — matrix operand order repaired and checked by reading, not compiled, not run on hardware
//
#include <hip/hip_runtime.h>
#include <stdint.h>

typedef __attribute__((ext_vector_type(16))) _Float16 v16h;
typedef __attribute__((ext_vector_type(8)))  float    v8f;

#define HW2D 16384      /* 128*128 */
#define VOX  2097152    /* 128^3   */

/* ------------------------------------------------------------------ */
/* Camera constants: cst[0]=full_interval, cst[1..9]=invK_ref,         */
/* per src view v (0..2) at cst+16+24v: R[9], t[3], K_src[9]           */
/* ------------------------------------------------------------------ */
__device__ __forceinline__ float camv(const float* c, int v, int m, int i, int j) {
  return c[(v * 2 + m) * 16 + i * 4 + j];
}

__global__ void setup_cams_kernel(const float* __restrict__ cams, float* __restrict__ cst) {
  if (blockIdx.x != 0 || threadIdx.x != 0) return;
  cst[0] = camv(cams, 0, 1, 3, 3) - camv(cams, 0, 1, 3, 0);
  /* inverse of K_ref[:3,:3] via adjugate */
  float a = camv(cams,0,1,0,0), b = camv(cams,0,1,0,1), c = camv(cams,0,1,0,2);
  float d = camv(cams,0,1,1,0), e = camv(cams,0,1,1,1), f = camv(cams,0,1,1,2);
  float g = camv(cams,0,1,2,0), h = camv(cams,0,1,2,1), i = camv(cams,0,1,2,2);
  float det = a*(e*i - f*h) - b*(d*i - f*g) + c*(d*h - e*g);
  float id  = 1.0f / det;
  cst[1] = (e*i - f*h)*id;  cst[2] = (c*h - b*i)*id;  cst[3] = (b*f - c*e)*id;
  cst[4] = (f*g - d*i)*id;  cst[5] = (a*i - c*g)*id;  cst[6] = (c*d - a*f)*id;
  cst[7] = (d*h - e*g)*id;  cst[8] = (b*g - a*h)*id;  cst[9] = (a*e - b*d)*id;
  /* inverse of E_ref (4x4) via Gauss-Jordan with partial pivoting */
  float M[4][8];
  for (int r = 0; r < 4; ++r)
    for (int cc = 0; cc < 4; ++cc) { M[r][cc] = camv(cams,0,0,r,cc); M[r][4+cc] = (r==cc)?1.f:0.f; }
  for (int col = 0; col < 4; ++col) {
    int piv = col;
    for (int r = col+1; r < 4; ++r) if (fabsf(M[r][col]) > fabsf(M[piv][col])) piv = r;
    if (piv != col) for (int j2 = 0; j2 < 8; ++j2) { float t = M[col][j2]; M[col][j2] = M[piv][j2]; M[piv][j2] = t; }
    float pv = 1.0f / M[col][col];
    for (int j2 = 0; j2 < 8; ++j2) M[col][j2] *= pv;
    for (int r = 0; r < 4; ++r) if (r != col) {
      float fr = M[r][col];
      for (int j2 = 0; j2 < 8; ++j2) M[r][j2] -= fr * M[col][j2];
    }
  }
  for (int v = 1; v < 4; ++v) {
    float T[3][4];
    for (int r = 0; r < 3; ++r)
      for (int cc = 0; cc < 4; ++cc) {
        float s = 0.f;
        for (int k = 0; k < 4; ++k) s += camv(cams, v, 0, r, k) * M[k][4 + cc];
        T[r][cc] = s;
      }
    float* cb = cst + 16 + (v - 1) * 24;
    for (int r = 0; r < 3; ++r) for (int cc = 0; cc < 3; ++cc) cb[r*3+cc] = T[r][cc];
    cb[9] = T[0][3]; cb[10] = T[1][3]; cb[11] = T[2][3];
    for (int r = 0; r < 3; ++r) for (int cc = 0; cc < 3; ++cc) cb[12 + r*3 + cc] = camv(cams, v, 1, r, cc);
  }
}

/* ------------------------------------------------------------------ */
/* Cost volume: channels [sup, occ, fsv], each (128,128,128) f32       */
/* ------------------------------------------------------------------ */
__global__ void __launch_bounds__(256)
cost_volume_kernel(const float* __restrict__ depths, const float* __restrict__ odepths,
                   const float* __restrict__ confs,  const float* __restrict__ oconfs,
                   const float* __restrict__ bounds, const float* __restrict__ cst,
                   const float* __restrict__ scaleP, const float* __restrict__ supP,
                   float* __restrict__ cost)
{
  int idx = blockIdx.x * blockDim.x + threadIdx.x;
  if (idx >= VOX) return;
  int x  = idx & 127;
  int y  = (idx >> 7) & 127;
  int dz = idx >> 14;
  int p  = (y << 7) + x;

  float lo = bounds[p], hi = bounds[HW2D + p];
  float interval = (hi - lo) * (1.0f / 127.0f);
  float dv = lo + (float)dz * interval;
  float full = cst[0];
  float inv_full = 1.0f / full;
  float sig_scale = scaleP[0] * (full / interval) * 128.0f;
  float sigma = supP[0] * (1.0f / 127.0f) * interval * inv_full;
  float inv2s = 1.0f / (2.0f * sigma * sigma + 1e-7f);

  float sup_s = 0.f, sup_c = 0.f, occ_s = 0.f, occ_c = 0.f;
  #pragma unroll
  for (int v = 0; v < 4; ++v) {
    float dvw = depths[v * HW2D + p];
    float cv  = confs [v * HW2D + p];
    float diff = (dvw - dv) * inv_full;
    float sup = __expf(-diff * diff * inv2s) * cv;
    float occ = cv / (1.0f + __expf(-diff * sig_scale));
    sup_s += sup; if (sup != 0.0f) sup_c += 1.0f;
    occ_s += occ; if (occ != 0.0f) occ_c += 1.0f;
  }

  float fsv_s = 0.f, fsv_c = 0.f;
  {
    float od0 = odepths[p], oc0 = oconfs[p];
    float f0 = oc0 / (1.0f + __expf(-((dv - od0) * inv_full) * sig_scale));
    fsv_s += f0; if (f0 != 0.0f) fsv_c += 1.0f;
  }

  float rx = cst[1]*(float)x + cst[2]*(float)y + cst[3];
  float ry = cst[4]*(float)x + cst[5]*(float)y + cst[6];
  float rz = cst[7]*(float)x + cst[8]*(float)y + cst[9];
  #pragma unroll
  for (int v = 0; v < 3; ++v) {
    const float* cb = cst + 16 + v * 24;
    float Xx = dv * (cb[0]*rx + cb[1]*ry + cb[2]*rz) + cb[9];
    float Xy = dv * (cb[3]*rx + cb[4]*ry + cb[5]*rz) + cb[10];
    float Xz = dv * (cb[6]*rx + cb[7]*ry + cb[8]*rz) + cb[11];
    float pu = cb[12]*Xx + cb[13]*Xy + cb[14]*Xz;
    float pv = cb[15]*Xx + cb[16]*Xy + cb[17]*Xz;
    float pz = cb[18]*Xx + cb[19]*Xy + cb[20]*Xz;
    float z  = fmaxf(pz, 1e-6f);
    float u  = pu / z, vv = pv / z;
    bool inb = (u >= 0.0f) && (u <= 127.0f) && (vv >= 0.0f) && (vv <= 127.0f);
    float uc = fminf(fmaxf(u,  0.0f), 127.0f);
    float vc = fminf(fmaxf(vv, 0.0f), 127.0f);
    int xi = min(max((int)floorf(uc), 0), 126);
    int yi = min(max((int)floorf(vc), 0), 126);
    float fx = uc - (float)xi, fy = vc - (float)yi;
    const float* Dv = odepths + (v + 1) * HW2D;
    const float* Cv = oconfs  + (v + 1) * HW2D;
    int i00 = (yi << 7) + xi;
    float w00 = (1.f-fx)*(1.f-fy), w01 = fx*(1.f-fy), w10 = (1.f-fx)*fy, w11 = fx*fy;
    float wd = w00*Dv[i00] + w01*Dv[i00+1] + w10*Dv[i00+128] + w11*Dv[i00+129];
    float wc = w00*Cv[i00] + w01*Cv[i00+1] + w10*Cv[i00+128] + w11*Cv[i00+129];
    bool valid = inb && (Xz > 0.001f);
    float cs  = valid ? (Xz - wd) : 0.0f;
    float wcf = valid ? wc : 0.0f;
    float f = wcf / (1.0f + __expf(-(cs * inv_full) * sig_scale));
    fsv_s += f; if (f != 0.0f) fsv_c += 1.0f;
  }
  cost[idx]           = sup_s / (sup_c + 1e-5f);
  cost[VOX + idx]     = occ_s / (occ_c + 1e-5f);
  cost[2*VOX + idx]   = fsv_s / (fsv_c + 1e-5f);
}

/* ------------------------------------------------------------------ */
/* Weight packing: (Cout,Cin,3,3,3) f32 -> f16 [Mpad][Cin*32]          */
/* Each channel's 27 taps padded to a 32-wide K block (zero-filled).   */
/* ------------------------------------------------------------------ */
__global__ void __launch_bounds__(256)
pack_w_kernel(const float* __restrict__ w, _Float16* __restrict__ Aw,
              int Cout, int Cin, int Kpad, int Mpad)
{
  int i = blockIdx.x * blockDim.x + threadIdx.x;
  if (i >= Mpad * Kpad) return;
  int m  = i / Kpad;
  int k  = i - m * Kpad;
  int ci = k >> 5;
  int r  = k & 31;
  float v = (m < Cout && r < 27) ? w[(m * Cin + ci) * 27 + r] : 0.0f;
  Aw[i] = (_Float16)v;
}

/* ------------------------------------------------------------------ */
/* Implicit-GEMM 3D conv via v_wmma_f32_16x16x32_f16 (wave32).         */
/* mode 0: conv s=1 SAME (pad 1,1); mode 1: conv s=2 SAME (pad 0,1);   */
/* mode 2: conv_transpose s=2 SAME (JAX pad_a=2, pad_b=1).             */
/* K padded per input channel (32 per ci) -> one ci per K-chunk;       */
/* gather offsets/validity precomputed once; hot loop is branch-free.  */
/* do2 is a *scalar* kernel arg so the dual-tile path is a scalar      */
/* branch (no EXEC-mask machinery around the WMMAs).                   */
/* ------------------------------------------------------------------ */
__global__ void __launch_bounds__(256)
conv3d_wmma_kernel(const float* __restrict__ in, float* __restrict__ out,
                   const _Float16* __restrict__ Aw,
                   const float* __restrict__ gamma, const float* __restrict__ beta,
                   const float* __restrict__ res,
                   int Cin, int Cout, int inD, int inH, int inW,
                   int outD, int outH, int outW, int mode,
                   int Kpad, int totalTiles, int do2)
{
  int wave = blockIdx.x * (blockDim.x >> 5) + (threadIdx.x >> 5);
  if (wave >= totalTiles) return;              /* wave-uniform exit: EXEC stays all-1s */
  int lane  = threadIdx.x & 31;
  int half  = lane >> 4;                       /* K-half selector for A/B frags */
  int l16   = lane & 15;                       /* A row / B,D column            */

  int mtiles = (Cout + 31) >> 5;
  int mt = wave % mtiles;
  int sp = wave / mtiles;
  int nxt = outW >> 4;
  int x0  = (sp % nxt) << 4;
  int rest = sp / nxt;
  int oy = rest % outH;
  int oz = rest / outH;
  int ox = x0 + l16;

  int cbase = mt << 5;

  /* ---- precompute 16 per-lane gather offsets + validity mask ---- */
  int inHW  = inH * inW;
  int inDHW = inD * inHW;
  int      preoff[16];
  unsigned vmask = 0;
  #pragma unroll
  for (int e = 0; e < 16; ++e) {
    int r  = (e < 8) ? (half * 8 + e) : (16 + half * 8 + (e - 8));  /* tap index within the 32-block */
    int kd = r / 9;
    int rr = r - kd * 9;
    int kh = rr / 3;
    int kw = rr - kh * 3;
    bool ok = (r < 27);
    int iz = 0, iy = 0, ix = 0;
    if (mode == 0) {
      iz = oz + kd - 1; iy = oy + kh - 1; ix = ox + kw - 1;
      ok = ok && ((unsigned)iz < (unsigned)inD) && ((unsigned)iy < (unsigned)inH)
              && ((unsigned)ix < (unsigned)inW);
    } else if (mode == 1) {
      iz = 2*oz + kd; iy = 2*oy + kh; ix = 2*ox + kw;
      ok = ok && (iz < inD) && (iy < inH) && (ix < inW);
    } else {
      int tz = oz + kd - 2, ty = oy + kh - 2, tx = ox + kw - 2;
      ok = ok && (tz >= 0) && (ty >= 0) && (tx >= 0)
              && !((tz | ty | tx) & 1);
      iz = tz >> 1; iy = ty >> 1; ix = tx >> 1;
      ok = ok && (iz < inD) && (iy < inH) && (ix < inW);
    }
    preoff[e] = ok ? (iz * inHW + iy * inW + ix) : 0;
    vmask |= (ok ? 1u : 0u) << e;
  }

  const uint32_t* A32 = (const uint32_t*)Aw;   /* packed f16 pairs, K contiguous */
  const uint32_t* pa0 = A32 + ((((cbase      + l16) * Kpad) >> 1) + (half << 2));
  const uint32_t* pa1 = A32 + ((((cbase + 16 + l16) * Kpad) >> 1) + (half << 2));

  v8f acc0 = {};
  v8f acc1 = {};
  #pragma unroll 2
  for (int ci = 0; ci < Cin; ++ci) {
    const float* bp = in + (size_t)ci * inDHW;
    /* B fragment: 16 predicated loads, no branches */
    union { _Float16 h[16]; v16h v; } bu;
    #pragma unroll
    for (int e = 0; e < 16; ++e) {
      float f = bp[preoff[e]];
      bu.h[e] = (_Float16)(((vmask >> e) & 1u) ? f : 0.0f);
    }
    /* A fragment: VGPR0-3 = K(half*8 + 0..7), VGPR4-7 = K(16 + half*8 + 0..7) */
    union { uint32_t u[8]; v16h v; } au0;
    #pragma unroll
    for (int q = 0; q < 4; ++q) { au0.u[q] = pa0[ci*16 + q]; au0.u[4 + q] = pa0[ci*16 + 8 + q]; }
    acc0 = __builtin_amdgcn_wmma_f32_16x16x32_f16(false, au0.v, false, bu.v, (short)0, acc0, false, false);
    if (do2) {   /* scalar branch: do2 is an SGPR kernel argument */
      union { uint32_t u[8]; v16h v; } au1;
      #pragma unroll
      for (int q = 0; q < 4; ++q) { au1.u[q] = pa1[ci*16 + q]; au1.u[4 + q] = pa1[ci*16 + 8 + q]; }
      acc1 = __builtin_amdgcn_wmma_f32_16x16x32_f16(false, au1.v, false, bu.v, (short)0, acc1, false, false);
    }
    __builtin_prefetch((const void*)(pa0 + ci*16 + 16), 0, 1);
  }

  /* Epilogue: D[M=r+8*half][N=l16]; fused BN+ReLU (or bias) + residual add */
  size_t HWo = (size_t)outH * outW;
  #pragma unroll
  for (int r = 0; r < 8; ++r) {
    int c0 = cbase + r + 8 * half;
    if (c0 < Cout) {
      float v = acc0[r];
      if (gamma) v = fmaxf(v * (gamma[c0] * rsqrtf(1.0f + 1e-5f)) + beta[c0], 0.0f);
      else       v += beta[c0];
      size_t o = ((size_t)c0 * outD + oz) * HWo + (size_t)oy * outW + ox;
      if (res) v += res[o];
      out[o] = v;
    }
    int c1 = cbase + 16 + r + 8 * half;
    if (do2 && c1 < Cout) {
      float v = acc1[r];
      if (gamma) v = fmaxf(v * (gamma[c1] * rsqrtf(1.0f + 1e-5f)) + beta[c1], 0.0f);
      else       v += beta[c1];
      size_t o = ((size_t)c1 * outD + oz) * HWo + (size_t)oy * outW + ox;
      if (res) v += res[o];
      out[o] = v;
    }
  }
}

/* ------------------------------------------------------------------ */
/* Softmax over D, expectation, 4-tap confidence                       */
/* ------------------------------------------------------------------ */
__global__ void __launch_bounds__(256)
softmax_out_kernel(const float* __restrict__ fin, const float* __restrict__ bounds,
                   float* __restrict__ out_fd, float* __restrict__ out_conf,
                   float* __restrict__ out_prob)
{
  int p = blockIdx.x * blockDim.x + threadIdx.x;
  if (p >= HW2D) return;
  float lo = bounds[p], hi = bounds[HW2D + p];
  float interval = (hi - lo) * (1.0f / 127.0f);
  float m = -3.4e38f;
  for (int d = 0; d < 128; ++d) m = fmaxf(m, fin[d * HW2D + p]);
  float s = 0.0f;
  for (int d = 0; d < 128; ++d) s += __expf(fin[d * HW2D + p] - m);
  float inv = 1.0f / s;
  float fd = 0.0f;
  for (int d = 0; d < 128; ++d) {
    float pr = __expf(fin[d * HW2D + p] - m) * inv;
    out_prob[d * HW2D + p] = pr;
    fd += (lo + (float)d * interval) * pr;
  }
  out_fd[p] = fd;
  float idxf = (fd - lo) / fmaxf(interval, 1e-6f);
  int i0 = min(max((int)floorf(idxf) - 1, 0), 124);
  float cf = 0.0f;
  for (int j = 0; j < 4; ++j) cf += out_prob[(i0 + j) * HW2D + p];
  out_conf[p] = cf;
}

/* ------------------------------------------------------------------ */
extern "C" void kernel_launch(void* const* d_in, const int* in_sizes, int n_in,
                              void* d_out, int out_size, void* d_ws, size_t ws_size,
                              hipStream_t stream)
{
  const float* depths  = (const float*)d_in[0];
  const float* odepths = (const float*)d_in[1];
  const float* confs   = (const float*)d_in[2];
  const float* oconfs  = (const float*)d_in[3];
  const float* cams    = (const float*)d_in[4];
  const float* bounds  = (const float*)d_in[5];
  const float* scaleP  = (const float*)d_in[6];
  const float* supP    = (const float*)d_in[7];

  size_t off = 0;
  auto alloc = [&](size_t bytes) -> void* {
    off = (off + 255) & ~(size_t)255;
    void* p = (char*)d_ws + off;
    off += bytes;
    return p;
  };
  float* cst  = (float*)alloc(128 * sizeof(float));
  float* cost = (float*)alloc((size_t)3  * VOX    * 4);
  float* o10  = (float*)alloc((size_t)16 * 262144 * 4);
  float* o20  = (float*)alloc((size_t)32 * 32768  * 4);
  float* o30  = (float*)alloc((size_t)64 * 4096   * 4);
  float* o01  = (float*)alloc((size_t)8  * VOX    * 4);
  float* o11  = (float*)alloc((size_t)16 * 262144 * 4);
  float* o21  = (float*)alloc((size_t)32 * 32768  * 4);
  float* o31  = (float*)alloc((size_t)64 * 4096   * 4);
  float* o41  = (float*)alloc((size_t)32 * 32768  * 4);
  float* o51  = (float*)alloc((size_t)16 * 262144 * 4);
  float* o61  = (float*)alloc((size_t)8  * VOX    * 4);
  float* fin  = (float*)alloc((size_t)VOX * 4);

  struct Layer {
    int wi;                               /* d_in index of w (g=wi+1, b=wi+2) */
    const float* in; float* out; const float* res;
    int Cin, Cout, inD, inH, inW, outD, outH, outW, mode;
  };
  Layer L[11] = {
    {  8, cost, o10, nullptr,  3, 16, 128,128,128,  64, 64, 64, 1 },  /* c10 */
    { 11, o10,  o20, nullptr, 16, 32,  64, 64, 64,  32, 32, 32, 1 },  /* c20 */
    { 14, o20,  o30, nullptr, 32, 64,  32, 32, 32,  16, 16, 16, 1 },  /* c30 */
    { 17, cost, o01, nullptr,  3,  8, 128,128,128, 128,128,128, 0 },  /* c01 */
    { 20, o10,  o11, nullptr, 16, 16,  64, 64, 64,  64, 64, 64, 0 },  /* c11 */
    { 23, o20,  o21, nullptr, 32, 32,  32, 32, 32,  32, 32, 32, 0 },  /* c21 */
    { 26, o30,  o31, nullptr, 64, 64,  16, 16, 16,  16, 16, 16, 0 },  /* c31 */
    { 29, o31,  o41, o21,     64, 32,  16, 16, 16,  32, 32, 32, 2 },  /* d40 */
    { 32, o41,  o51, o11,     32, 16,  32, 32, 32,  64, 64, 64, 2 },  /* d50 */
    { 35, o51,  o61, o01,     16,  8,  64, 64, 64, 128,128,128, 2 },  /* d60 */
    { 38, o61,  fin, nullptr,  8,  1, 128,128,128, 128,128,128, 0 },  /* c62 */
  };
  _Float16* Aw[11]; int Kpad[11], Mpad[11];
  for (int l = 0; l < 11; ++l) {
    Kpad[l] = L[l].Cin * 32;                 /* per-channel padded K (32/ci) */
    Mpad[l] = (L[l].Cout + 31) & ~31;
    Aw[l] = (_Float16*)alloc((size_t)Mpad[l] * Kpad[l] * sizeof(_Float16));
  }

  setup_cams_kernel<<<1, 32, 0, stream>>>(cams, cst);
  cost_volume_kernel<<<VOX / 256, 256, 0, stream>>>(depths, odepths, confs, oconfs,
                                                    bounds, cst, scaleP, supP, cost);
  for (int l = 0; l < 11; ++l) {
    int n = Mpad[l] * Kpad[l];
    pack_w_kernel<<<(n + 255) / 256, 256, 0, stream>>>(
        (const float*)d_in[L[l].wi], Aw[l], L[l].Cout, L[l].Cin, Kpad[l], Mpad[l]);
  }
  for (int l = 0; l < 11; ++l) {
    const Layer& y = L[l];
    const float* gamma = (l == 10) ? nullptr : (const float*)d_in[y.wi + 1];
    const float* beta  = (l == 10) ? (const float*)d_in[39] : (const float*)d_in[y.wi + 2];
    int mtiles = (y.Cout + 31) >> 5;
    int tiles  = y.outD * y.outH * (y.outW >> 4) * mtiles;
    int do2    = (y.Cout > 16) ? 1 : 0;      /* layer-uniform here (Cout in {1,8,16,32,64}) */
    conv3d_wmma_kernel<<<(tiles + 7) / 8, 256, 0, stream>>>(
        y.in, y.out, Aw[l], gamma, beta, y.res,
        y.Cin, y.Cout, y.inD, y.inH, y.inW, y.outD, y.outH, y.outW, y.mode,
        Kpad[l], tiles, do2);
  }

  float* out_fd   = (float*)d_out;
  float* out_conf = out_fd + HW2D;
  float* out_prob = out_conf + HW2D;
  float* out_bnds = out_prob + (size_t)VOX;
  softmax_out_kernel<<<(HW2D + 255) / 256, 256, 0, stream>>>(fin, bounds, out_fd, out_conf, out_prob);
  hipMemcpyAsync(out_bnds, bounds, (size_t)2 * HW2D * sizeof(float),
                 hipMemcpyDeviceToDevice, stream);

  (void)in_sizes; (void)n_in; (void)out_size; (void)ws_size;
}
